// HetNet_34857954574962
// MI455X (gfx1250) — matrix operand order repaired and checked by reading, not compiled, not run on hardware
//
// HetNet for MI455X / gfx1250 — wave32 + v_wmma_f32_16x16x32_f16 (f32 accum).
// Fuses every 3-layer D=32 MLP + its mean-reduction into one kernel (the 128MB
// [B,Ns,F,D] intermediates never exist). Layers run in "transposed" orientation
// (H^T = W^T X^T) so the inter-layer C->B relayout is 2x ds_store_b128 +
// 2x ds_load_b128. Hot kernels pipeline TWO independent 16-row tiles per
// iteration so the WMMA->VALU RAW hazard (4 co-exec slots) is filled with the
// other tile's work. Inter-layer ReLU is done packed-f16 (v_pk_max_num_f16)
// after the cvt_pk, halving the VALU in the exchange path.
#include <hip/hip_runtime.h>

typedef __attribute__((ext_vector_type(16))) _Float16 v16h;
typedef __attribute__((ext_vector_type(8)))  _Float16 v8h;
typedef __attribute__((ext_vector_type(8)))  float    v8f;

#define DEV static __device__ __forceinline__

DEV float relu(float x) { return fmaxf(x, 0.0f); }

// f32x8 -> f16x8 with ReLU applied in packed f16 (cvt order is equivalent:
// negative f32 -> negative/-0 f16, then max(+0, .) clamps identically).
DEV v8h cvt_relu8(v8f c) {
  v8h r;
#pragma unroll
  for (int v = 0; v < 8; ++v) r[v] = (_Float16)c[v];
  const v8h z = {};
  return __builtin_elementwise_max(r, z);
}

DEV v8f wmma32(v16h a, v16h b, v8f c) {
  // D = A(16x32 f16) * B(32x16 f16) + C(16x16 f32)
  return __builtin_amdgcn_wmma_f32_16x16x32_f16(false, a, false, b, (short)0, c,
                                                false, false);
}

// A-fragment of W^T. W row-major [Ktot x 32]; chunk picks output cols 16c..16c+15.
// A order used here (contiguous k, matches B): lane m=L%16; elem e -> k = e + 16*(L>=16).
DEV v16h load_wT_frag(const float* __restrict__ W, int kbase, int chunk) {
  const int lane = threadIdx.x & 31;
  const int m = chunk * 16 + (lane & 15);
  const int hi8 = (lane >> 4) << 3;
  v16h a;
#pragma unroll
  for (int e = 0; e < 16; ++e) {
    int k = kbase + e + 2 * hi8;
    a[e] = (_Float16)W[k * 32 + m];
  }
  return a;
}

// C-fragment init from bias: lane holds (f = v + 8*hi + 16*chunk, n = L%16).
DEV v8f load_bias_cfrag(const float* __restrict__ b, int chunk) {
  const float* p = b + chunk * 16 + (((threadIdx.x & 31) >> 4) << 3);
  v8f c;
#pragma unroll
  for (int v = 0; v < 8; ++v) c[v] = p[v];
  return c;
}

// B-fragment from a 32-float f32 row (row chosen per-lane): b[e] = row[16*hi + e].
DEV v16h load_row_bfrag(const float* __restrict__ rowp) {
  const float* p = rowp + 16 * (((threadIdx.x & 31) >> 4));
  v16h bf;
#pragma unroll
  for (int e = 0; e < 16; ++e) bf[e] = (_Float16)p[e];
  return bf;
}

// ReLU(C0,C1) -> f16 -> per-wave LDS tile (16 rows x 40 halfs, 80B stride) -> next
// B-fragment. Same-wave DS ops are in-order, so no barrier is needed.
DEV v16h relu_exchange(v8f c0, v8f c1, _Float16* tile) {
  const int lane = threadIdx.x & 31;
  const int n = lane & 15, hi = lane >> 4;
  v8h r0 = cvt_relu8(c0);
  v8h r1 = cvt_relu8(c1);
  _Float16* row = tile + n * 40;
  *(v8h*)(row + 8 * hi)      = r0;  // feats 8hi..8hi+7
  *(v8h*)(row + 16 + 8 * hi) = r1;  // feats 16+8hi..16+8hi+7
  v8h lo = *(v8h*)(row + 16 * hi);
  v8h hh = *(v8h*)(row + 16 * hi + 8);
  v16h bf;
#pragma unroll
  for (int e = 0; e < 8; ++e) { bf[e] = lo[e]; bf[e + 8] = hh[e]; }
  return bf;
}

// ---------- Stage 1: pv over scalar inputs, summed over the set axis (Ns) -------
__global__ void __launch_bounds__(128) pv_sum_kernel(
    const float* __restrict__ src, int N, int F,
    const float* __restrict__ W1, const float* __restrict__ b1,
    const float* __restrict__ W2, const float* __restrict__ b2,
    const float* __restrict__ W3, const float* __restrict__ b3,
    float* __restrict__ out_sum) {
  __shared__ __attribute__((aligned(16))) _Float16 tile[8][640];
  const int wave = threadIdx.x >> 5, lane = threadIdx.x & 31;
  const int n = lane & 15, hi = lane >> 4;
  const int job = blockIdx.x * 4 + wave;   // (b, f)
  const int b = job / F, f = job % F;
  _Float16* ta = tile[wave * 2];
  _Float16* tb = tile[wave * 2 + 1];

  float w0l[16], b0l[16];
#pragma unroll
  for (int e = 0; e < 16; ++e) { w0l[e] = W1[16 * hi + e]; b0l[e] = b1[16 * hi + e]; }
  v16h A2[2], A3[2]; v8f cb2[2], cb3[2];
#pragma unroll
  for (int c = 0; c < 2; ++c) {
    A2[c] = load_wT_frag(W2, 0, c); A3[c] = load_wT_frag(W3, 0, c);
    cb2[c] = load_bias_cfrag(b2, c); cb3[c] = load_bias_cfrag(b3, c);
  }
  v8f p0 = {}, p1 = {};
  const float* sb = src + (long long)b * N * F + f;
  for (int s0 = 0; s0 < N; s0 += 32) {      // two independent 16-row tiles
    float xa = sb[(long long)(s0 + n) * F];
    float xb = sb[(long long)(s0 + 16 + n) * F];
    v8f t0a, t1a, t0b, t1b;
#pragma unroll
    for (int e = 0; e < 8; ++e) {           // layer1 (1->32) in VALU
      t0a[e] = fmaf(xa, w0l[e], b0l[e]);
      t1a[e] = fmaf(xa, w0l[e + 8], b0l[e + 8]);
      t0b[e] = fmaf(xb, w0l[e], b0l[e]);
      t1b[e] = fmaf(xb, w0l[e + 8], b0l[e + 8]);
    }
    v8h h0a = cvt_relu8(t0a), h1a = cvt_relu8(t1a);
    v8h h0b = cvt_relu8(t0b), h1b = cvt_relu8(t1b);
    v16h Ba, Bb;
#pragma unroll
    for (int e = 0; e < 8; ++e) {
      Ba[e] = h0a[e]; Ba[e + 8] = h1a[e];
      Bb[e] = h0b[e]; Bb[e + 8] = h1b[e];
    }
    v8f c0a = wmma32(A2[0], Ba, cb2[0]);
    v8f c1a = wmma32(A2[1], Ba, cb2[1]);
    v8f c0b = wmma32(A2[0], Bb, cb2[0]);
    v8f c1b = wmma32(A2[1], Bb, cb2[1]);
    v16h B3a = relu_exchange(c0a, c1a, ta);
    v16h B3b = relu_exchange(c0b, c1b, tb);
    c0a = wmma32(A3[0], B3a, cb3[0]);
    c1a = wmma32(A3[1], B3a, cb3[1]);
    c0b = wmma32(A3[0], B3b, cb3[0]);
    c1b = wmma32(A3[1], B3b, cb3[1]);
#pragma unroll
    for (int v = 0; v < 8; ++v) {
      p0[v] += relu(c0a[v]) + relu(c0b[v]);
      p1[v] += relu(c1a[v]) + relu(c1b[v]);
    }
  }
#pragma unroll
  for (int m = 1; m < 16; m <<= 1)
#pragma unroll
    for (int v = 0; v < 8; ++v) {
      p0[v] += __shfl_xor(p0[v], m, 32);
      p1[v] += __shfl_xor(p1[v], m, 32);
    }
  if (n == 0) {
    float* o = out_sum + (long long)job * 32 + 8 * hi;
#pragma unroll
    for (int v = 0; v < 8; ++v) { o[v] = p0[v]; o[16 + v] = p1[v]; }
  }
}

// ---------- Generic 3x(32->32) ReLU MLP on rows: in = sA*A[r] (+ sB*B[r]) -------
__global__ void __launch_bounds__(128) rows_mlp3_kernel(
    const float* __restrict__ inA, const float* __restrict__ inB, float sA, float sB,
    const float* __restrict__ W1, const float* __restrict__ b1,
    const float* __restrict__ W2, const float* __restrict__ b2,
    const float* __restrict__ W3, const float* __restrict__ b3,
    float* __restrict__ out) {
  __shared__ __attribute__((aligned(16))) _Float16 tile[4][640];
  __shared__ __attribute__((aligned(16))) float otile[4][576];
  const int wave = threadIdx.x >> 5, lane = threadIdx.x & 31;
  const int n = lane & 15, hi = lane >> 4;
  const int r0 = (blockIdx.x * 4 + wave) * 16;
  _Float16* mytile = tile[wave];
  float* mot = otile[wave];

  v16h A1[2], A2[2], A3[2]; v8f cb1[2], cb2[2], cb3[2];
#pragma unroll
  for (int c = 0; c < 2; ++c) {
    A1[c] = load_wT_frag(W1, 0, c); A2[c] = load_wT_frag(W2, 0, c);
    A3[c] = load_wT_frag(W3, 0, c);
    cb1[c] = load_bias_cfrag(b1, c); cb2[c] = load_bias_cfrag(b2, c);
    cb3[c] = load_bias_cfrag(b3, c);
  }
  const float* pa = inA + (long long)(r0 + n) * 32 + 16 * hi;
  v16h B1;
  if (inB) {
    const float* pb = inB + (long long)(r0 + n) * 32 + 16 * hi;
#pragma unroll
    for (int e = 0; e < 16; ++e) B1[e] = (_Float16)fmaf(sA, pa[e], sB * pb[e]);
  } else {
#pragma unroll
    for (int e = 0; e < 16; ++e) B1[e] = (_Float16)(sA * pa[e]);
  }
  v8f c0 = wmma32(A1[0], B1, cb1[0]);
  v8f c1 = wmma32(A1[1], B1, cb1[1]);
  v16h B2 = relu_exchange(c0, c1, mytile);
  c0 = wmma32(A2[0], B2, cb2[0]);
  c1 = wmma32(A2[1], B2, cb2[1]);
  v16h B3 = relu_exchange(c0, c1, mytile);
  c0 = wmma32(A3[0], B3, cb3[0]);
  c1 = wmma32(A3[1], B3, cb3[1]);
#pragma unroll
  for (int v = 0; v < 8; ++v) {              // ReLU + f32 transpose tile
    mot[n * 36 + 8 * hi + v]      = relu(c0[v]);
    mot[n * 36 + 16 + 8 * hi + v] = relu(c1[v]);
  }
  const float* rrow = mot + n * 36 + 16 * hi;
  float* og = out + (long long)(r0 + n) * 32 + 16 * hi;
#pragma unroll
  for (int e = 0; e < 16; ++e) og[e] = rrow[e];
}

// ---------- [scalar, vec32] -> puf/pfv/pfz MLP, summed over reduction axis -----
// mode 0: jobs over N (s or q), reduce over f (scalar stride 1, vec = buf[b,f,:])
// mode 1: jobs over F,          reduce over s (scalar stride F, vec = buf[b,s,:])
__global__ void __launch_bounds__(128) concat_mlp_sum_kernel(
    const float* __restrict__ src, int N, int F, int mode,
    const float* __restrict__ vecbuf,
    const float* __restrict__ W1, const float* __restrict__ b1,  // [33,32]
    const float* __restrict__ W2, const float* __restrict__ b2,
    const float* __restrict__ W3, const float* __restrict__ b3,
    float* __restrict__ out_sum) {
  __shared__ __attribute__((aligned(16))) _Float16 tile[8][640];
  const int wave = threadIdx.x >> 5, lane = threadIdx.x & 31;
  const int n = lane & 15, hi = lane >> 4;
  const int J = mode ? F : N, RED = mode ? N : F;
  const int job = blockIdx.x * 4 + wave;
  const int b = job / J, j = job % J;
  _Float16* ta = tile[wave * 2];
  _Float16* tb = tile[wave * 2 + 1];

  const long long sbase = mode ? ((long long)b * N * F + j) : (((long long)b * N + j) * F);
  const long long rstride = mode ? F : 1;
  const float* vb = vecbuf + (long long)b * RED * 32;

  v16h A1v[2], A2[2], A3[2]; v8f cb1[2], cb2[2], cb3[2];
  float w0c[2][8];                              // scalar column (row 0 of W1)
#pragma unroll
  for (int c = 0; c < 2; ++c) {
    A1v[c] = load_wT_frag(W1 + 32, 0, c);       // rows 1..32 (vec features)
    A2[c]  = load_wT_frag(W2, 0, c);
    A3[c]  = load_wT_frag(W3, 0, c);
    cb1[c] = load_bias_cfrag(b1, c); cb2[c] = load_bias_cfrag(b2, c);
    cb3[c] = load_bias_cfrag(b3, c);
#pragma unroll
    for (int v = 0; v < 8; ++v) w0c[c][v] = W1[v + 8 * hi + 16 * c];
  }
  v8f p0 = {}, p1 = {};
  for (int rr = 0; rr < RED; rr += 32) {        // two independent 16-row tiles
    const int ra = rr + n, rb = rr + 16 + n;
    const float xa = src[sbase + (long long)ra * rstride];
    const float xb = src[sbase + (long long)rb * rstride];
    v16h Bva = load_row_bfrag(vb + (long long)ra * 32);
    v16h Bvb = load_row_bfrag(vb + (long long)rb * 32);
    v8f c0a = wmma32(A1v[0], Bva, cb1[0]);
    v8f c1a = wmma32(A1v[1], Bva, cb1[1]);
    v8f c0b = wmma32(A1v[0], Bvb, cb1[0]);
    v8f c1b = wmma32(A1v[1], Bvb, cb1[1]);
#pragma unroll
    for (int v = 0; v < 8; ++v) {               // + x * W1[0,:] (lane owns its row)
      c0a[v] = fmaf(xa, w0c[0][v], c0a[v]);
      c1a[v] = fmaf(xa, w0c[1][v], c1a[v]);
      c0b[v] = fmaf(xb, w0c[0][v], c0b[v]);
      c1b[v] = fmaf(xb, w0c[1][v], c1b[v]);
    }
    v16h B2a = relu_exchange(c0a, c1a, ta);
    v16h B2b = relu_exchange(c0b, c1b, tb);
    c0a = wmma32(A2[0], B2a, cb2[0]);
    c1a = wmma32(A2[1], B2a, cb2[1]);
    c0b = wmma32(A2[0], B2b, cb2[0]);
    c1b = wmma32(A2[1], B2b, cb2[1]);
    v16h B3a = relu_exchange(c0a, c1a, ta);
    v16h B3b = relu_exchange(c0b, c1b, tb);
    c0a = wmma32(A3[0], B3a, cb3[0]);
    c1a = wmma32(A3[1], B3a, cb3[1]);
    c0b = wmma32(A3[0], B3b, cb3[0]);
    c1b = wmma32(A3[1], B3b, cb3[1]);
#pragma unroll
    for (int v = 0; v < 8; ++v) {
      p0[v] += relu(c0a[v]) + relu(c0b[v]);
      p1[v] += relu(c1a[v]) + relu(c1b[v]);
    }
  }
#pragma unroll
  for (int m = 1; m < 16; m <<= 1)
#pragma unroll
    for (int v = 0; v < 8; ++v) {
      p0[v] += __shfl_xor(p0[v], m, 32);
      p1[v] += __shfl_xor(p1[v], m, 32);
    }
  if (n == 0) {
    float* o = out_sum + (long long)job * 32 + 8 * hi;
#pragma unroll
    for (int v = 0; v < 8; ++v) { o[v] = p0[v]; o[16 + v] = p1[v]; }
  }
}

// ---------- Head: pfy([z, in_ys]) : 64->32->32->1 (final linear) ---------------
__global__ void __launch_bounds__(128) head_kernel(
    const float* __restrict__ zbuf, const float* __restrict__ ybuf, int Nq, int Fy,
    const float* __restrict__ W1, const float* __restrict__ b1,   // [64,32]
    const float* __restrict__ W2, const float* __restrict__ b2,
    const float* __restrict__ W3, const float* __restrict__ b3,   // [32,1],[1]
    float* __restrict__ out) {
  __shared__ __attribute__((aligned(16))) _Float16 tile[8][640];
  const int wave = threadIdx.x >> 5, lane = threadIdx.x & 31;
  const int n = lane & 15, hi = lane >> 4;
  const int job = blockIdx.x * 4 + wave;  // b*Nq + q
  const int b = job / Nq;
  _Float16* ta = tile[wave * 2];
  _Float16* tb = tile[wave * 2 + 1];

  v16h A1z[2], A1y[2], A2[2]; v8f cb1[2], cb2[2];
  float w3l[2][8];
#pragma unroll
  for (int c = 0; c < 2; ++c) {
    A1z[c] = load_wT_frag(W1, 0, c);    // k = 0..31  (z part)
    A1y[c] = load_wT_frag(W1, 32, c);   // k = 32..63 (in_ys part)
    A2[c]  = load_wT_frag(W2, 0, c);
    cb1[c] = load_bias_cfrag(b1, c); cb2[c] = load_bias_cfrag(b2, c);
#pragma unroll
    for (int v = 0; v < 8; ++v) w3l[c][v] = W3[v + 8 * hi + 16 * c];
  }
  const float bias3 = b3[0];
  const v16h Bz = load_row_bfrag(zbuf + (long long)job * 32);  // same for all rows
  const float* yb = ybuf + (long long)b * Fy * 32;
  float* ob = out + (long long)job * Fy;

  for (int f0 = 0; f0 < Fy; f0 += 32) {   // two independent 16-row tiles
    v16h Bya = load_row_bfrag(yb + (long long)(f0 + n) * 32);
    v16h Byb = load_row_bfrag(yb + (long long)(f0 + 16 + n) * 32);
    v8f c0a = wmma32(A1z[0], Bz, cb1[0]);
    c0a = wmma32(A1y[0], Bya, c0a);
    v8f c1a = wmma32(A1z[1], Bz, cb1[1]);
    c1a = wmma32(A1y[1], Bya, c1a);
    v8f c0b = wmma32(A1z[0], Bz, cb1[0]);
    c0b = wmma32(A1y[0], Byb, c0b);
    v8f c1b = wmma32(A1z[1], Bz, cb1[1]);
    c1b = wmma32(A1y[1], Byb, c1b);
    v16h B2a = relu_exchange(c0a, c1a, ta);
    v16h B2b = relu_exchange(c0b, c1b, tb);
    c0a = wmma32(A2[0], B2a, cb2[0]);
    c1a = wmma32(A2[1], B2a, cb2[1]);
    c0b = wmma32(A2[0], B2b, cb2[0]);
    c1b = wmma32(A2[1], B2b, cb2[1]);
    float ty = 0.f, tz = 0.f;             // 32->1 final linear (no ReLU)
#pragma unroll
    for (int v = 0; v < 8; ++v) {
      ty = fmaf(relu(c0a[v]), w3l[0][v], ty);
      ty = fmaf(relu(c1a[v]), w3l[1][v], ty);
      tz = fmaf(relu(c0b[v]), w3l[0][v], tz);
      tz = fmaf(relu(c1b[v]), w3l[1][v], tz);
    }
    ty += __shfl_xor(ty, 16, 32);         // combine feature halves
    tz += __shfl_xor(tz, 16, 32);
    if (hi == 0) {
      ob[f0 + n] = ty + bias3;
      ob[f0 + 16 + n] = tz + bias3;
    }
  }
}

extern "C" void kernel_launch(void* const* d_in, const int* in_sizes, int n_in,
                              void* d_out, int out_size, void* d_ws, size_t ws_size,
                              hipStream_t stream) {
  (void)in_sizes; (void)n_in; (void)out_size; (void)ws_size;
  const int B = 8, Ns = 1024, Nq = 1024, Fx = 128, Fy = 128;
  auto g = [&](int i) { return (const float*)d_in[i]; };
  const float* que_x = g(0);
  const float* sup_x = g(1);
  const float* sup_y = g(2);
  const int pv = 3, pc = 9, puf = 15, pug = 21, pfv = 27, pgv = 33, pfz = 39,
            pgz = 45, pfy = 51;

  float* ws = (float*)d_ws;
  size_t o = 0;
  float* vs_sum = ws + o; o += (size_t)B * Fx * 32;
  float* cs_sum = ws + o; o += (size_t)B * Fy * 32;
  float* vs     = ws + o; o += (size_t)B * Fx * 32;
  float* cs     = ws + o; o += (size_t)B * Fy * 32;
  float* uxs    = ws + o; o += (size_t)B * Ns * 32;
  float* uys    = ws + o; o += (size_t)B * Ns * 32;
  float* u_s    = ws + o; o += (size_t)B * Ns * 32;
  float* inxs   = ws + o; o += (size_t)B * Fx * 32;
  float* inys   = ws + o; o += (size_t)B * Fy * 32;
  float* in_x   = ws + o; o += (size_t)B * Fx * 32;
  float* in_y   = ws + o; o += (size_t)B * Fy * 32;
  float* z_sum  = ws + o; o += (size_t)B * Nq * 32;
  float* zb     = ws + o; o += (size_t)B * Nq * 32;

  const dim3 blk(128);
  // vs / cs : pv (sum over Ns) then pc (with 1/Ns)
  pv_sum_kernel<<<B * Fx / 4, blk, 0, stream>>>(sup_x, Ns, Fx, g(pv), g(pv + 1),
      g(pv + 2), g(pv + 3), g(pv + 4), g(pv + 5), vs_sum);
  pv_sum_kernel<<<B * Fy / 4, blk, 0, stream>>>(sup_y, Ns, Fy, g(pv), g(pv + 1),
      g(pv + 2), g(pv + 3), g(pv + 4), g(pv + 5), cs_sum);
  rows_mlp3_kernel<<<B * Fx / 64, blk, 0, stream>>>(vs_sum, nullptr, 1.f / Ns, 0.f,
      g(pc), g(pc + 1), g(pc + 2), g(pc + 3), g(pc + 4), g(pc + 5), vs);
  rows_mlp3_kernel<<<B * Fy / 64, blk, 0, stream>>>(cs_sum, nullptr, 1.f / Ns, 0.f,
      g(pc), g(pc + 1), g(pc + 2), g(pc + 3), g(pc + 4), g(pc + 5), cs);
  // u stage: puf (sum over F), then pug on (u_xs/Fx + u_ys/Fy)
  concat_mlp_sum_kernel<<<B * Ns / 4, blk, 0, stream>>>(sup_x, Ns, Fx, 0, vs,
      g(puf), g(puf + 1), g(puf + 2), g(puf + 3), g(puf + 4), g(puf + 5), uxs);
  concat_mlp_sum_kernel<<<B * Ns / 4, blk, 0, stream>>>(sup_y, Ns, Fy, 0, cs,
      g(puf), g(puf + 1), g(puf + 2), g(puf + 3), g(puf + 4), g(puf + 5), uys);
  rows_mlp3_kernel<<<B * Ns / 64, blk, 0, stream>>>(uxs, uys, 1.f / Fx, 1.f / Fy,
      g(pug), g(pug + 1), g(pug + 2), g(pug + 3), g(pug + 4), g(pug + 5), u_s);
  // in stage: pfv (sum over Ns), then pgv (with 1/Ns)
  concat_mlp_sum_kernel<<<B * Fx / 4, blk, 0, stream>>>(sup_x, Ns, Fx, 1, u_s,
      g(pfv), g(pfv + 1), g(pfv + 2), g(pfv + 3), g(pfv + 4), g(pfv + 5), inxs);
  concat_mlp_sum_kernel<<<B * Fy / 4, blk, 0, stream>>>(sup_y, Ns, Fy, 1, u_s,
      g(pfv), g(pfv + 1), g(pfv + 2), g(pfv + 3), g(pfv + 4), g(pfv + 5), inys);
  rows_mlp3_kernel<<<B * Fx / 64, blk, 0, stream>>>(inxs, nullptr, 1.f / Ns, 0.f,
      g(pgv), g(pgv + 1), g(pgv + 2), g(pgv + 3), g(pgv + 4), g(pgv + 5), in_x);
  rows_mlp3_kernel<<<B * Fy / 64, blk, 0, stream>>>(inys, nullptr, 1.f / Ns, 0.f,
      g(pgv), g(pgv + 1), g(pgv + 2), g(pgv + 3), g(pgv + 4), g(pgv + 5), in_y);
  // z stage: pfz (sum over Fx), then pgz (with 1/Fx)
  concat_mlp_sum_kernel<<<B * Nq / 4, blk, 0, stream>>>(que_x, Nq, Fx, 0, in_x,
      g(pfz), g(pfz + 1), g(pfz + 2), g(pfz + 3), g(pfz + 4), g(pfz + 5), z_sum);
  rows_mlp3_kernel<<<B * Nq / 64, blk, 0, stream>>>(z_sum, nullptr, 1.f / Fx, 0.f,
      g(pgz), g(pgz + 1), g(pgz + 2), g(pgz + 3), g(pgz + 4), g(pgz + 5), zb);
  // head: pfy -> d_out [B,Nq,Fy] (reference reshape is a flat view)
  head_kernel<<<B * Nq / 4, blk, 0, stream>>>(zb, in_y, Nq, Fy, g(pfy), g(pfy + 1),
      g(pfy + 2), g(pfy + 3), g(pfy + 4), g(pfy + 5), (float*)d_out);
}
